// PointFeaturePropagation_34943853920281
// MI455X (gfx1250) — matrix operand order repaired and checked by reading, not compiled, not run on hardware
//
#include <hip/hip_runtime.h>

// ---------------------------------------------------------------------------
// CDNA5 (gfx1250, wave32) fused PointNet++ feature propagation
//   Kernel 1: pack fp32 weights -> bf16 WMMA B-fragment layout (once, tiny)
//   Kernel 2: KNN(k=3) + inverse-distance interpolation (LDS-staged xyz2)
//   Kernel 3: fused 3-layer MLP via v_wmma_f32_16x16x32_bf16
// ---------------------------------------------------------------------------

typedef __attribute__((ext_vector_type(16))) __bf16 v16bf;
typedef __attribute__((ext_vector_type(8)))  float  v8f;

struct __align__(16) U128 { unsigned int x, y, z, w; };
struct __align__(16) F4   { float x, y, z, w; };
union  Frag { v16bf v; U128 q[2]; };

#define BATCH 4
#define NP1   16384
#define NP2   4096
#define ROWS  (BATCH * NP1)     // 65536
#define CIN   320               // 64 feats1 + 256 interp
#define P1LD  67                // 3 + 64
#define P2LD  259               // 3 + 256
#define OUTLD 131               // 3 + 128

// float -> bf16 bits, round-to-nearest-even
__device__ __forceinline__ unsigned short f2bf(float f) {
    unsigned int u = __float_as_uint(f);
    u += 0x7FFFu + ((u >> 16) & 1u);
    return (unsigned short)(u >> 16);
}

// A-fragment (16x32 bf16, row-major source, ld in elements).
// lane<16:  row=lane,    elems = K[k0..k0+7] ++ K[k0+16..k0+23]
// lane>=16: row=lane-16, elems = K[k0+8..k0+15] ++ K[k0+24..k0+31]
__device__ __forceinline__ v16bf load_a(const unsigned short* base, int ld,
                                        int kbase, int lane) {
    const int row = lane & 15;
    const int k0  = kbase + ((lane & 16) ? 8 : 0);
    const unsigned short* p = base + row * ld + k0;
    Frag f;
    f.q[0] = *(const U128*)(p);
    f.q[1] = *(const U128*)(p + 16);
    return f.v;
}

// B-fragment from pre-packed weights: tile (kk,nt) stores 32 lanes x 16 bf16
// contiguously -> one 32B read per lane.
__device__ __forceinline__ v16bf load_b(const unsigned short* wp, int NT,
                                        int kk, int nt, int lane) {
    const unsigned short* p = wp + ((size_t)(((kk * NT) + nt) * 32 + lane) << 4);
    Frag f;
    f.q[0] = ((const U128*)p)[0];
    f.q[1] = ((const U128*)p)[1];
    return f.v;
}

// ---------------------------------------------------------------------------
// Weight packing: W (K x N, fp32 row-major) -> bf16 fragment tiles.
// Element t = (((kt*NT + nt)*32 + lane)*16 + e):
//   lane<16:  k = kt*32 + (e<8 ? e : e+8),     n = nt*16 + lane
//   lane>=16: k = kt*32 + 8 + (e<8 ? e : e+8), n = nt*16 + lane - 16
// ---------------------------------------------------------------------------
__global__ __launch_bounds__(256) void pack_w_kernel(
    const float* __restrict__ W, unsigned short* __restrict__ Wp, int K, int N)
{
    const int t = blockIdx.x * 256 + threadIdx.x;
    if (t >= K * N) return;
    const int e    = t & 15;
    const int lane = (t >> 4) & 31;
    const int rest = t >> 9;
    const int NT   = N >> 4;
    const int nt   = rest % NT;
    const int kt   = rest / NT;
    const int k    = kt * 32 + ((lane & 16) ? 8 : 0) + (e & 7) + ((e & 8) ? 16 : 0);
    const int n    = nt * 16 + (lane & 15);
    Wp[t] = f2bf(W[(size_t)k * N + n]);
}

// ---------------------------------------------------------------------------
// KNN (k=3) + interpolation. One thread per point1. xyz2 + |xyz2|^2 in LDS.
// Rank by s = bb - 2*dot (monotone in d^2). Writes:
//   out[row*131 + 0..2]  = xyz1 (fp32)
//   h0[row*320 + 0..319] = [feats1 | interp] (bf16 bits)
// ---------------------------------------------------------------------------
__global__ __launch_bounds__(256) void knn_interp_kernel(
    const float* __restrict__ p1, const float* __restrict__ p2,
    unsigned short* __restrict__ h0, float* __restrict__ out)
{
    __shared__ F4 ref[NP2];   // 64 KB
    const int b = blockIdx.y;
    const int n = blockIdx.x * 256 + threadIdx.x;
    const float* p2b = p2 + (size_t)b * NP2 * P2LD;

    for (int j = threadIdx.x; j < NP2; j += 256) {
        const float* q = p2b + (size_t)j * P2LD;
        F4 r; r.x = q[0]; r.y = q[1]; r.z = q[2];
        r.w = fmaf(r.x, r.x, fmaf(r.y, r.y, r.z * r.z));
        ref[j] = r;
    }
    __syncthreads();

    const float* p1r = p1 + ((size_t)b * NP1 + n) * P1LD;
    const float x1 = p1r[0], y1 = p1r[1], z1 = p1r[2];

    float s0 = 3.4e38f, s1 = 3.4e38f, s2 = 3.4e38f;
    int   i0 = 0, i1 = 0, i2 = 0;
    for (int j = 0; j < NP2; ++j) {
        const F4 r = ref[j];
        const float dot = fmaf(x1, r.x, fmaf(y1, r.y, z1 * r.z));
        const float s   = fmaf(-2.f, dot, r.w);
        if (s < s0)      { s2 = s1; i2 = i1; s1 = s0; i1 = i0; s0 = s; i0 = j; }
        else if (s < s1) { s2 = s1; i2 = i1; s1 = s;  i1 = j; }
        else if (s < s2) { s2 = s;  i2 = j; }
    }
    const float aa = fmaf(x1, x1, fmaf(y1, y1, z1 * z1));
    const float d0 = sqrtf(fmaxf(aa + s0, 0.f));
    const float d1 = sqrtf(fmaxf(aa + s1, 0.f));
    const float d2 = sqrtf(fmaxf(aa + s2, 0.f));
    float w0 = 1.f / (d0 + 1e-8f);
    float w1 = 1.f / (d1 + 1e-8f);
    float w2 = 1.f / (d2 + 1e-8f);
    const float inv = 1.f / (w0 + w1 + w2);
    w0 *= inv; w1 *= inv; w2 *= inv;

    const size_t row = (size_t)b * NP1 + n;
    float* orow = out + row * OUTLD;
    orow[0] = x1; orow[1] = y1; orow[2] = z1;

    unsigned short* hrow = h0 + row * CIN;
    #pragma unroll 8
    for (int c = 0; c < 64; ++c) hrow[c] = f2bf(p1r[3 + c]);

    const float* g0 = p2b + (size_t)i0 * P2LD + 3;
    const float* g1 = p2b + (size_t)i1 * P2LD + 3;
    const float* g2 = p2b + (size_t)i2 * P2LD + 3;
    __builtin_prefetch(g0, 0, 3);
    __builtin_prefetch(g1, 0, 3);
    __builtin_prefetch(g2, 0, 3);
    #pragma unroll 8
    for (int c = 0; c < 256; ++c) {
        const float v = w0 * g0[c] + w1 * g1[c] + w2 * g2[c];
        hrow[64 + c] = f2bf(v);
    }
}

// ---------------------------------------------------------------------------
// Fused MLP: [64x320]@[320x256] relu -> @[256x128] relu -> @[128x128] relu.
// Block = 256 threads = 8 waves: wave = (m-tile 0..3, n-half 0..1).
// bf16 WMMA with f32 accumulate; intermediates in LDS as bf16.
// ---------------------------------------------------------------------------
__global__ __launch_bounds__(256) void mlp_fused_kernel(
    const unsigned short* __restrict__ h0,
    const unsigned short* __restrict__ W1p, const float* __restrict__ b1,
    const unsigned short* __restrict__ W2p, const float* __restrict__ b2,
    const unsigned short* __restrict__ W3p, const float* __restrict__ b3,
    float* __restrict__ out)
{
    __shared__ __align__(16) unsigned short h1[64 * 256];  // 32 KB
    __shared__ __align__(16) unsigned short h2[64 * 128];  // 16 KB

    const int lane = threadIdx.x & 31;
    const int wave = threadIdx.x >> 5;
    const int mt   = wave & 3;        // m-tile within block
    const int nh   = wave >> 2;       // n-half
    const int rowbase = blockIdx.x * 64 + mt * 16;
    const int ncol = lane & 15;                 // D col = lane%16
    const int mrow = (lane & 16) ? 8 : 0;       // D rows: 0-7 or 8-15

    // ---- Layer 1: K=320 (10 k-steps), N=256 (16 n-tiles, 8 per wave) ----
    v16bf a1[10];
    const unsigned short* A0 = h0 + (size_t)rowbase * CIN;
    #pragma unroll
    for (int kk = 0; kk < 10; ++kk) a1[kk] = load_a(A0, CIN, kk * 32, lane);

    #pragma unroll
    for (int ni = 0; ni < 8; ++ni) {
        const int nt = nh * 8 + ni;
        v8f acc = {0.f, 0.f, 0.f, 0.f, 0.f, 0.f, 0.f, 0.f};
        #pragma unroll
        for (int kk = 0; kk < 10; ++kk) {
            const v16bf bf = load_b(W1p, 16, kk, nt, lane);
            acc = __builtin_amdgcn_wmma_f32_16x16x32_bf16(
                false, a1[kk], false, bf, (short)0, acc, false, false);
        }
        const int nc = nt * 16 + ncol;
        const float bias = b1[nc];
        #pragma unroll
        for (int r = 0; r < 8; ++r) {
            const float v = fmaxf(acc[r] + bias, 0.f);
            h1[(mt * 16 + mrow + r) * 256 + nc] = f2bf(v);
        }
    }
    __syncthreads();

    // ---- Layer 2: K=256 (8 k-steps), N=128 (8 n-tiles, 4 per wave) ----
    v16bf a2[8];
    #pragma unroll
    for (int kk = 0; kk < 8; ++kk) a2[kk] = load_a(&h1[mt * 16 * 256], 256, kk * 32, lane);

    #pragma unroll
    for (int ni = 0; ni < 4; ++ni) {
        const int nt = nh * 4 + ni;
        v8f acc = {0.f, 0.f, 0.f, 0.f, 0.f, 0.f, 0.f, 0.f};
        #pragma unroll
        for (int kk = 0; kk < 8; ++kk) {
            const v16bf bf = load_b(W2p, 8, kk, nt, lane);
            acc = __builtin_amdgcn_wmma_f32_16x16x32_bf16(
                false, a2[kk], false, bf, (short)0, acc, false, false);
        }
        const int nc = nt * 16 + ncol;
        const float bias = b2[nc];
        #pragma unroll
        for (int r = 0; r < 8; ++r) {
            const float v = fmaxf(acc[r] + bias, 0.f);
            h2[(mt * 16 + mrow + r) * 128 + nc] = f2bf(v);
        }
    }
    __syncthreads();

    // ---- Layer 3: K=128 (4 k-steps), N=128 (8 n-tiles, 4 per wave) ----
    v16bf a3[4];
    #pragma unroll
    for (int kk = 0; kk < 4; ++kk) a3[kk] = load_a(&h2[mt * 16 * 128], 128, kk * 32, lane);

    #pragma unroll
    for (int ni = 0; ni < 4; ++ni) {
        const int nt = nh * 4 + ni;
        v8f acc = {0.f, 0.f, 0.f, 0.f, 0.f, 0.f, 0.f, 0.f};
        #pragma unroll
        for (int kk = 0; kk < 4; ++kk) {
            const v16bf bf = load_b(W3p, 8, kk, nt, lane);
            acc = __builtin_amdgcn_wmma_f32_16x16x32_bf16(
                false, a3[kk], false, bf, (short)0, acc, false, false);
        }
        const int nc = nt * 16 + ncol;
        const float bias = b3[nc];
        #pragma unroll
        for (int r = 0; r < 8; ++r) {
            const float v = fmaxf(acc[r] + bias, 0.f);
            out[(size_t)(rowbase + mrow + r) * OUTLD + 3 + nc] = v;
        }
    }
}

// ---------------------------------------------------------------------------
extern "C" void kernel_launch(void* const* d_in, const int* in_sizes, int n_in,
                              void* d_out, int out_size, void* d_ws, size_t ws_size,
                              hipStream_t stream) {
    const float* p1 = (const float*)d_in[0];  // (4,16384,67)
    const float* p2 = (const float*)d_in[1];  // (4,4096,259)
    const float* W1 = (const float*)d_in[2];  // (320,256)
    const float* b1 = (const float*)d_in[3];  // (256)
    const float* W2 = (const float*)d_in[4];  // (256,128)
    const float* b2 = (const float*)d_in[5];  // (128)
    const float* W3 = (const float*)d_in[6];  // (128,128)
    const float* b3 = (const float*)d_in[7];  // (128)
    float* out = (float*)d_out;               // (4,16384,131)

    // Workspace carving (all 16B aligned):
    unsigned short* h0  = (unsigned short*)d_ws;          // 65536*320 bf16 = 40 MB
    unsigned short* W1p = h0  + (size_t)ROWS * CIN;       // 320*256 bf16
    unsigned short* W2p = W1p + 320 * 256;                // 256*128 bf16
    unsigned short* W3p = W2p + 256 * 128;                // 128*128 bf16

    pack_w_kernel<<<(320 * 256 + 255) / 256, 256, 0, stream>>>(W1, W1p, 320, 256);
    pack_w_kernel<<<(256 * 128 + 255) / 256, 256, 0, stream>>>(W2, W2p, 256, 128);
    pack_w_kernel<<<(128 * 128 + 255) / 256, 256, 0, stream>>>(W3, W3p, 128, 128);

    dim3 gA(NP1 / 256, BATCH);
    knn_interp_kernel<<<gA, 256, 0, stream>>>(p1, p2, h0, out);

    mlp_fused_kernel<<<ROWS / 64, 256, 0, stream>>>(h0, W1p, b1, W2p, b2, W3p, b3, out);
}
